// ImprovedHybridColorGenerator_60309930771207
// MI455X (gfx1250) — compile-verified, matrix operand order checked by reading, not run on hardware
//
#include <hip/hip_runtime.h>

// ============================================================================
// MI455X (gfx1250) implementation of ImprovedHybridColorGenerator forward.
//
// Roofline: ~520 GFLOP/call (65536 samples x ~4M MAC). Weights ~8MB f16 (L2
// resident). Activation HBM traffic ~1.5GB => ~65us @ 23.3TB/s. Matmul-bound:
// v_wmma_f32_16x16x32_f16 (f16 in, f32 acc), BN folded into weights,
// activation epilogues fused and fully specialized at compile time.
// wave32: each wave computes a 16(M) x 64(N) tile per N-chunk, K step 32.
// GEMM is templated on <KP, N, ACT>: no runtime branches anywhere in the
// hot loop or epilogue.
// ============================================================================

typedef __attribute__((ext_vector_type(16))) _Float16 v16h;
typedef __attribute__((ext_vector_type(8)))  _Float16 v8h;
typedef __attribute__((ext_vector_type(8)))  float    v8f;

#define EPSBN 1e-5f

__device__ __forceinline__ float clampf(float v, float lo, float hi) {
  return fminf(fmaxf(v, lo), hi);
}
__device__ __forceinline__ float sigmoidf_(float v) {
  return 1.0f / (1.0f + __expf(-v));
}

// ---------------------------------------------------------------------------
// Fold BN into linear: Wt[n][k] = W[k][n] * s_n  (f16, transposed, K zero-
// padded to Kp, N zero-padded to Np), bias[n] = (b-m)*s + be.
// ---------------------------------------------------------------------------
__global__ void fold_kernel(const float* __restrict__ W, const float* __restrict__ b,
                            const float* __restrict__ g, const float* __restrict__ be,
                            const float* __restrict__ m, const float* __restrict__ v,
                            _Float16* __restrict__ Wt, float* __restrict__ bias,
                            int K, int N, int Kp, int Np)
{
  int n = blockIdx.x * blockDim.x + threadIdx.x;
  if (n >= Np) return;
  if (n >= N) {
    bias[n] = 0.f;
    for (int k = 0; k < Kp; ++k) Wt[(size_t)n * Kp + k] = (_Float16)0.f;
    return;
  }
  float s = 1.f, off = 0.f;
  if (g) { s = g[n] * rsqrtf(v[n] + EPSBN); off = be[n] - m[n] * s; }
  bias[n] = b[n] * s + off;
  for (int k = 0; k < Kp; ++k) {
    float w = (k < K) ? W[(size_t)k * N + n] * s : 0.f;
    Wt[(size_t)n * Kp + k] = (_Float16)w;
  }
}

// ---------------------------------------------------------------------------
// WMMA GEMM, fully specialized: out = ACT(A[M,KP] @ Wt[N,KP]^T + bias).
// ACT: 1 = lrelu -> f16 out, 2 = sigmoid -> f32 out, 3 = +resid, lrelu -> f16.
// Block: 256 thr = 8 waves; wave w owns rows (blk*8+w)*16 .. +15.
// Weight chunk (64 x KP) staged in LDS with +8h row pad (bank spread).
// A/B fragment per-lane layout per CDNA5 ISA 16-bit A-matrix table; B uses
// the mirrored layout via the transposed weight store.
// ---------------------------------------------------------------------------
template <int KP, int N, int ACT>
__global__ __launch_bounds__(256) void gemm_kernel(
    const _Float16* __restrict__ A,   // [M, KP]
    const _Float16* __restrict__ Wt,  // [N, KP] folded transposed weights
    const float*    __restrict__ bias,// [N]
    const _Float16* __restrict__ resid, // [M, N] (ACT==3 only)
    _Float16*       __restrict__ outH,  // [M, N] (ACT==1/3)
    float*          __restrict__ outF)  // [M, N] (ACT==2)
{
  constexpr int PITCH = KP + 8;                 // halfs per LDS row
  constexpr int KV    = KP / 8;                 // v8h per weight row
  extern __shared__ _Float16 Bs[];              // [64][PITCH]
  const int lane = threadIdx.x & 31;
  const int wave = threadIdx.x >> 5;
  const int l15  = lane & 15;
  const int kg   = (lane >> 4) << 3;            // 0 or 8: K-group base
  const int rowA  = (blockIdx.x * 8 + wave) * 16 + l15;
  const int crow0 = (blockIdx.x * 8 + wave) * 16 + ((lane >> 4) << 3);
  const _Float16* aRow = A + (size_t)rowA * KP;

#pragma unroll 1
  for (int nbase = 0; nbase < N; nbase += 64) {
    { // cooperative global -> LDS stage of the 64 x KP weight chunk
      const _Float16* src = Wt + (size_t)nbase * KP;
#pragma unroll
      for (int it = 0; it < KP / 32; ++it) {    // 8*KP v8h units / 256 thr
        int u  = it * 256 + threadIdx.x;
        int rr = u / KV;                        // compile-time divisor
        int kk = (u - rr * KV) << 3;
        *(v8h*)(Bs + rr * PITCH + kk) = *(const v8h*)(src + (size_t)rr * KP + kk);
      }
    }
    __syncthreads();

    v8f z = {0.f,0.f,0.f,0.f,0.f,0.f,0.f,0.f};
    v8f acc0 = z, acc1 = z, acc2 = z, acc3 = z;

#pragma unroll 2
    for (int kb = 0; kb < KP; kb += 32) {
      v16h a;
      *((v8h*)&a)     = *(const v8h*)(aRow + kb + kg);
      *((v8h*)&a + 1) = *(const v8h*)(aRow + kb + kg + 16);
      const _Float16* bp = Bs + l15 * PITCH + kb + kg;
      v16h b0, b1, b2, b3;
      *((v8h*)&b0)     = *(const v8h*)(bp);
      *((v8h*)&b0 + 1) = *(const v8h*)(bp + 16);
      *((v8h*)&b1)     = *(const v8h*)(bp + 16 * PITCH);
      *((v8h*)&b1 + 1) = *(const v8h*)(bp + 16 * PITCH + 16);
      *((v8h*)&b2)     = *(const v8h*)(bp + 32 * PITCH);
      *((v8h*)&b2 + 1) = *(const v8h*)(bp + 32 * PITCH + 16);
      *((v8h*)&b3)     = *(const v8h*)(bp + 48 * PITCH);
      *((v8h*)&b3 + 1) = *(const v8h*)(bp + 48 * PITCH + 16);
      acc0 = __builtin_amdgcn_wmma_f32_16x16x32_f16(false, a, false, b0, (short)0, acc0, false, false);
      acc1 = __builtin_amdgcn_wmma_f32_16x16x32_f16(false, a, false, b1, (short)0, acc1, false, false);
      acc2 = __builtin_amdgcn_wmma_f32_16x16x32_f16(false, a, false, b2, (short)0, acc2, false, false);
      acc3 = __builtin_amdgcn_wmma_f32_16x16x32_f16(false, a, false, b3, (short)0, acc3, false, false);
    }

    v8f accs[4] = {acc0, acc1, acc2, acc3};
#pragma unroll
    for (int t = 0; t < 4; ++t) {
      const int col = nbase + t * 16 + l15;
      const float bv = bias[col];
#pragma unroll
      for (int i = 0; i < 8; ++i) {
        const size_t idx = (size_t)(crow0 + i) * N + col;
        float v = accs[t][i] + bv;
        if (ACT == 3) v += (float)resid[idx];       // compile-time branch
        if (ACT == 1 || ACT == 3) v = v >= 0.f ? v : 0.2f * v;
        if (ACT == 2) v = sigmoidf_(v);
        if (ACT == 2) outF[idx] = v;
        else          outH[idx] = (_Float16)v;
      }
    }
    __syncthreads();
  }
}

template <int KP, int N, int ACT>
static void run_gemm(int Bn, hipStream_t stream,
                     const _Float16* A, const _Float16* Wt, const float* bias,
                     const _Float16* resid, _Float16* oh, float* of)
{
  const size_t lds = (size_t)64 * (KP + 8) * sizeof(_Float16);
  gemm_kernel<KP, N, ACT><<<Bn / 128, 256, lds, stream>>>(A, Wt, bias, resid, oh, of);
}

// ---------------------------------------------------------------------------
// Elementwise helpers
// ---------------------------------------------------------------------------
__global__ void pack_x_kernel(const float* __restrict__ x, _Float16* __restrict__ A0, int Bn)
{
  int r = blockIdx.x * blockDim.x + threadIdx.x;
  if (r >= Bn) return;
#pragma unroll
  for (int k = 0; k < 32; ++k)
    A0[(size_t)r * 32 + k] = (_Float16)((k < 18) ? x[(size_t)r * 18 + k] : 0.f);
}

__global__ void theme3_kernel(const _Float16* __restrict__ t2,
    const float* __restrict__ W, const float* __restrict__ b,
    float* __restrict__ isl, int Bn)
{
  int r = blockIdx.x * blockDim.x + threadIdx.x;
  if (r >= Bn) return;
  float s = b[0];
  const _Float16* tr = t2 + (size_t)r * 64;     // T2 stored with stride 64
#pragma unroll
  for (int k = 0; k < 32; ++k) s += (float)tr[k] * W[k];
  isl[r] = (sigmoidf_(s) > 0.5f) ? 1.f : 0.f;
}

__global__ void copy_feats_kernel(const _Float16* __restrict__ fe,
                                  _Float16* __restrict__ cin, int Bn)
{
  size_t t = (size_t)blockIdx.x * blockDim.x + threadIdx.x;
  size_t total = (size_t)Bn * 48;               // v8h units of 384 cols
  if (t >= total) return;
  size_t row = t / 48, i = t - row * 48;
  *(v8h*)(cin + row * 480 + i * 8) = *(const v8h*)(fe + row * 384 + i * 8);
}

// base stage-0 colors per reference _forward
__device__ __forceinline__ void compute_s0(const float* xr, bool il, int isDark,
                                           float dbg, float lt, float tn, float* s0)
{
  for (int j = 0; j < 6; ++j) {
    int mod = j % 3;
    int aj  = (j < 3) ? 2 : 5;                  // ACCENT_IDX
    for (int c = 0; c < 3; ++c) {
      float v = xr[j * 3 + c];
      float f;
      if (!isDark) {
        if (mod == 0)      f = 0.9f - 0.2f * (1.f - v);
        else if (mod == 1) f = 0.1f + 0.2f * v;
        else               f = clampf(v * 0.9f, 0.f, 1.f);
        s0[j * 3 + c] = il ? v : f;
      } else {
        if (mod == 0)      f = clampf(dbg + 0.05f * (1.f - v) + tn * xr[aj * 3 + c], 0.f, 1.f);
        else if (mod == 1) f = lt + 0.15f * (1.f - v);
        else               f = clampf(v * 1.3f, 0.f, 1.f);
        s0[j * 3 + c] = il ? f : v;
      }
    }
  }
}

// writes cols 384..479 of refiner input: [x(18) | base54 | zeros24]
__global__ void build_tail_kernel(const float* __restrict__ x,
    const float* __restrict__ isl, const float* __restrict__ dbg_p,
    const float* __restrict__ lt_p, const float* __restrict__ tn_p,
    _Float16* __restrict__ cin, int isDark, int Bn)
{
  int r = blockIdx.x * blockDim.x + threadIdx.x;
  if (r >= Bn) return;
  float xr[18];
#pragma unroll
  for (int k = 0; k < 18; ++k) xr[k] = x[(size_t)r * 18 + k];
  float dbg = clampf(dbg_p[0], 0.01f, 0.4f);
  float lt  = clampf(lt_p[0], 0.6f, 0.99f);
  float tn  = clampf(tn_p[0], 0.0f, 0.3f);
  float s0[18];
  compute_s0(xr, isl[r] > 0.5f, isDark, dbg, lt, tn, s0);
  _Float16* t = cin + (size_t)r * 480;
#pragma unroll
  for (int k = 0; k < 18; ++k) t[384 + k] = (_Float16)xr[k];
#pragma unroll
  for (int k = 0; k < 54; ++k) t[402 + k] = (_Float16)0.f;
#pragma unroll
  for (int j = 0; j < 6; ++j)
    for (int c = 0; c < 3; ++c) t[402 + j * 9 + c] = (_Float16)s0[j * 3 + c];
#pragma unroll
  for (int k = 456; k < 480; ++k) t[k] = (_Float16)0.f;
}

// reference _rules, one thread per row; refiner out padded to stride 192
__global__ void rules_kernel(const float* __restrict__ ro, const float* __restrict__ x,
    const float* __restrict__ isl, const float* __restrict__ alg_w,
    const float* __restrict__ mc_p, const float* __restrict__ dbg_p,
    const float* __restrict__ lt_p, const float* __restrict__ tn_p,
    float* __restrict__ out, int isDark, int Bn)
{
  int r = blockIdx.x * blockDim.x + threadIdx.x;
  if (r >= Bn) return;
  float xr[18];
#pragma unroll
  for (int k = 0; k < 18; ++k) xr[k] = x[(size_t)r * 18 + k];
  bool il = isl[r] > 0.5f;
  float dbg = clampf(dbg_p[0], 0.01f, 0.4f);
  float lt  = clampf(lt_p[0], 0.6f, 0.99f);
  float tn  = clampf(tn_p[0], 0.0f, 0.3f);
  float s0[18];
  compute_s0(xr, il, isDark, dbg, lt, tn, s0);

  const float* rr = ro + (size_t)r * 192;
  bool match = isDark ? il : !il;
  float mc = clampf(mc_p[0], 0.3f, 0.7f);
  float st0[18];
  for (int j = 0; j < 6; ++j) {
    float w = clampf(sigmoidf_(alg_w[j]), 0.2f, 0.8f) * (match ? 1.f : 0.5f);
    for (int c = 0; c < 3; ++c)
      st0[j * 3 + c] = w * s0[j * 3 + c] + (1.f - w) * rr[j * 27 + c];
  }
  const float LR = 0.299f, LG = 0.587f, LB = 0.114f;
  for (int i = 1; i <= 4; i += 3) {
    float bg = st0[(i-1)*3]*LR + st0[(i-1)*3+1]*LG + st0[(i-1)*3+2]*LB;
    float tl = st0[i*3]*LR + st0[i*3+1]*LG + st0[i*3+2]*LB;
    bool need = fabsf(bg - tl) < mc;
    bool darken = isDark ? (bg >= 0.5f) : (bg < 0.5f);
    if (need)
      for (int c = 0; c < 3; ++c) {
        float cur = st0[i*3+c];
        st0[i*3+c] = darken ? clampf(cur * 0.7f, 0.f, 1.f)
                            : clampf(cur + 0.3f, 0.f, 1.f);
      }
  }
  float* o = out + (size_t)r * 324 + (isDark ? 162 : 0);
  for (int j = 0; j < 6; ++j)
    for (int s = 0; s < 9; ++s)
      for (int c = 0; c < 3; ++c) {
        int idx = j * 27 + s * 3 + c;
        o[idx] = (s == 0) ? st0[j * 3 + c] : rr[idx];
      }
}

// ---------------------------------------------------------------------------
// Host orchestration. Input flattening assumption: d_in[0]=x, then params
// pytree-flattened with sorted dict keys at every level (JAX convention):
// alg_w, dark(50), dark_bg, enc(42), light(50), light_text, mc_dark,
// mc_light, theme(14), tint.  Refiner leaf layout (base r): bn1[4] bn2[4]
// l1{W,b} l2{W,b} l3{W,b} r1[12] r2[12] r3[12]; res leaf layout: bn1[4]
// bn2[4] l1{W,b} l2{W,b}; bn leaf order: be,g,m,v.
// ---------------------------------------------------------------------------
extern "C" void kernel_launch(void* const* d_in, const int* in_sizes, int n_in,
                              void* d_out, int out_size, void* d_ws, size_t ws_size,
                              hipStream_t stream)
{
  (void)n_in; (void)out_size; (void)ws_size;
  const int Bn = in_sizes[0] / 18;
  const float* x          = (const float*)d_in[0];
  const float* alg_w      = (const float*)d_in[1];
  const float* dark_bg    = (const float*)d_in[52];
  const float* light_text = (const float*)d_in[145];
  const float* mc_dark    = (const float*)d_in[146];
  const float* mc_light   = (const float*)d_in[147];
  const float* themeW3    = (const float*)d_in[160];
  const float* themeB3    = (const float*)d_in[161];
  const float* tint       = (const float*)d_in[162];
  float* out = (float*)d_out;

  struct Lin { int wi, bi, bni, K, N, Kp, Np; };
  Lin L[27];
  L[0] = {156,157,148, 18, 64, 32, 64};   // theme.l1
  L[1] = {158,159,152, 64, 32, 64, 64};   // theme.l2 (N pad 64)
  L[2] = { 65, 66, 53, 18,128, 32,128};   // enc.l1
  L[3] = { 79, 80, 71,128,128,128,128};   // enc.r1.l1
  L[4] = { 81, 82, 75,128,128,128,128};   // enc.r1.l2
  L[5] = { 67, 68, 57,128,256,128,256};   // enc.l2
  L[6] = { 91, 92, 83,256,256,256,256};   // enc.r2.l1
  L[7] = { 93, 94, 87,256,256,256,256};   // enc.r2.l2
  L[8] = { 69, 70, 61,256,384,256,384};   // enc.l3
  int li = 9;
  for (int s = 0; s < 2; ++s) {
    int r = (s == 0) ? 2 : 95;            // dark params at 2.., light at 95..
    L[li++] = {r+ 8, r+ 9, r+ 0, 456,512,480,512}; // l1
    L[li++] = {r+22, r+23, r+14, 512,512,512,512}; // r1.l1
    L[li++] = {r+24, r+25, r+18, 512,512,512,512}; // r1.l2
    L[li++] = {r+34, r+35, r+26, 512,512,512,512}; // r2.l1
    L[li++] = {r+36, r+37, r+30, 512,512,512,512}; // r2.l2
    L[li++] = {r+10, r+11, r+ 4, 512,384,512,384}; // l2
    L[li++] = {r+46, r+47, r+38, 384,384,384,384}; // r3.l1
    L[li++] = {r+48, r+49, r+42, 384,384,384,384}; // r3.l2
    L[li++] = {r+12, r+13, -1,   384,162,384,192}; // l3 (no BN, N pad 192)
  }

  size_t off = 0;
  auto alloc = [&](size_t bytes) -> void* {
    size_t a = (off + 255) & ~(size_t)255;
    off = a + bytes;
    return (void*)((char*)d_ws + a);
  };

  _Float16* wt[27]; float* bb[27];
  for (int i = 0; i < 27; ++i) {
    wt[i] = (_Float16*)alloc((size_t)L[i].Np * L[i].Kp * sizeof(_Float16));
    bb[i] = (float*)alloc((size_t)L[i].Np * sizeof(float));
  }
  _Float16* A0   = (_Float16*)alloc((size_t)Bn * 32  * 2);
  _Float16* P    = (_Float16*)alloc((size_t)Bn * 512 * 2);
  _Float16* Q    = (_Float16*)alloc((size_t)Bn * 512 * 2);
  _Float16* Ra   = (_Float16*)alloc((size_t)Bn * 384 * 2);
  _Float16* Rb   = (_Float16*)alloc((size_t)Bn * 384 * 2);
  _Float16* FE   = (_Float16*)alloc((size_t)Bn * 384 * 2);
  _Float16* CIN  = (_Float16*)alloc((size_t)Bn * 480 * 2);
  float*    ROUT = (float*)   alloc((size_t)Bn * 192 * 4);
  float*    ISL  = (float*)   alloc((size_t)Bn * 4);

  // fold BN into all linears (tiny kernels)
  for (int i = 0; i < 27; ++i) {
    const float* Wp = (const float*)d_in[L[i].wi];
    const float* bp = (const float*)d_in[L[i].bi];
    const float *g = nullptr, *be = nullptr, *m = nullptr, *v = nullptr;
    if (L[i].bni >= 0) {
      be = (const float*)d_in[L[i].bni + 0];
      g  = (const float*)d_in[L[i].bni + 1];
      m  = (const float*)d_in[L[i].bni + 2];
      v  = (const float*)d_in[L[i].bni + 3];
    }
    fold_kernel<<<(L[i].Np + 63) / 64, 64, 0, stream>>>(
        Wp, bp, g, be, m, v, wt[i], bb[i], L[i].K, L[i].N, L[i].Kp, L[i].Np);
  }

  pack_x_kernel<<<(Bn + 255) / 256, 256, 0, stream>>>(x, A0, Bn);

  // theme head: 18->64->32(pad 64) (WMMA), final 32->1 as scalar dot
  run_gemm< 32, 64, 1>(Bn, stream, A0, wt[0], bb[0], nullptr, Q, nullptr);
  run_gemm< 64, 64, 1>(Bn, stream, Q,  wt[1], bb[1], nullptr, P, nullptr);
  theme3_kernel<<<(Bn + 255) / 256, 256, 0, stream>>>(P, themeW3, themeB3, ISL, Bn);

  // encoder -> feats[B,384]
  run_gemm< 32,128, 1>(Bn, stream, A0, wt[2], bb[2], nullptr, P,  nullptr); // 18->128
  run_gemm<128,128, 1>(Bn, stream, P,  wt[3], bb[3], nullptr, Q,  nullptr); // r1.l1
  run_gemm<128,128, 3>(Bn, stream, Q,  wt[4], bb[4], P,       P,  nullptr); // r1.l2+res
  run_gemm<128,256, 1>(Bn, stream, P,  wt[5], bb[5], nullptr, Ra, nullptr); // 128->256
  run_gemm<256,256, 1>(Bn, stream, Ra, wt[6], bb[6], nullptr, Rb, nullptr); // r2.l1
  run_gemm<256,256, 3>(Bn, stream, Rb, wt[7], bb[7], Ra,      Ra, nullptr); // r2.l2+res
  run_gemm<256,384, 1>(Bn, stream, Ra, wt[8], bb[8], nullptr, FE, nullptr); // 256->384

  // refiners: side 0 = dark (out cols 162..323), side 1 = light (0..161)
  for (int side = 0; side < 2; ++side) {
    const int isDark = (side == 0) ? 1 : 0;
    const int b0     = (side == 0) ? 9 : 18;
    const float* mc  = isDark ? mc_dark : mc_light;
    copy_feats_kernel<<<(int)(((size_t)Bn * 48 + 255) / 256), 256, 0, stream>>>(FE, CIN, Bn);
    build_tail_kernel<<<(Bn + 255) / 256, 256, 0, stream>>>(
        x, ISL, dark_bg, light_text, tint, CIN, isDark, Bn);
    run_gemm<480,512, 1>(Bn, stream, CIN, wt[b0+0], bb[b0+0], nullptr, P,  nullptr);
    run_gemm<512,512, 1>(Bn, stream, P,   wt[b0+1], bb[b0+1], nullptr, Q,  nullptr);
    run_gemm<512,512, 3>(Bn, stream, Q,   wt[b0+2], bb[b0+2], P,       P,  nullptr);
    run_gemm<512,512, 1>(Bn, stream, P,   wt[b0+3], bb[b0+3], nullptr, Q,  nullptr);
    run_gemm<512,512, 3>(Bn, stream, Q,   wt[b0+4], bb[b0+4], P,       P,  nullptr);
    run_gemm<512,384, 1>(Bn, stream, P,   wt[b0+5], bb[b0+5], nullptr, Ra, nullptr);
    run_gemm<384,384, 1>(Bn, stream, Ra,  wt[b0+6], bb[b0+6], nullptr, Rb, nullptr);
    run_gemm<384,384, 3>(Bn, stream, Rb,  wt[b0+7], bb[b0+7], Ra,      Ra, nullptr);
    run_gemm<384,192, 2>(Bn, stream, Ra,  wt[b0+8], bb[b0+8], nullptr, nullptr, ROUT);
    rules_kernel<<<(Bn + 255) / 256, 256, 0, stream>>>(
        ROUT, x, ISL, alg_w, mc, dark_bg, light_text, tint, out, isDark, Bn);
  }
}
